// Head_10239202034003
// MI455X (gfx1250) — matrix operand order repaired and checked by reading, not compiled, not run on hardware
//
#include <hip/hip_runtime.h>
#include <hip/hip_bf16.h>
#include <stdint.h>

typedef __bf16 bf16_t;
typedef __attribute__((ext_vector_type(16))) __bf16 v16bf;
typedef __attribute__((ext_vector_type(8)))  float  v8f;
typedef __attribute__((ext_vector_type(4))) unsigned int u32x4;
typedef __attribute__((ext_vector_type(8)))  int    i32x8;
typedef __attribute__((ext_vector_type(4)))  int    i32x4;

#define DD   1024      // n_embd
#define HS   64        // head size
#define SEQ  2048
#define NB   8
#define BS   (NB*SEQ)  // 16384 rows
#define WSZ  (HS*DD)   // 65536 elements per weight matrix

// ---------------------------------------------------------------------------
// Kernel 0: one-time f32 -> bf16 conversion of the three weight matrices.
// Removes per-tile weight conversion VALU + halves L2 weight traffic.
// ---------------------------------------------------------------------------
__global__ __launch_bounds__(256) void wcvt(
    const float* __restrict__ Wk, const float* __restrict__ Wq,
    const float* __restrict__ Wv, bf16_t* __restrict__ Wb)
{
    const int i = blockIdx.x * 256 + threadIdx.x;   // 0..65535
    Wb[i]            = (bf16_t)Wq[i];
    Wb[WSZ + i]      = (bf16_t)Wk[i];
    Wb[2 * WSZ + i]  = (bf16_t)Wv[i];
}

// ---------------------------------------------------------------------------
// Kernel 1: QKV projection.  y[s,h] = sum_d X[s,d] * W[h,d]
// One wave per 16-row tile of X; v_wmma_f32_16x16x32_bf16 over D=1024.
// Q pre-scaled by 1/sqrt(D)=1/32.  V stored transposed per batch: Vt[b][h][s].
// ---------------------------------------------------------------------------
__global__ __launch_bounds__(128) void qkv_proj(
    const float* __restrict__ X,
    const bf16_t* __restrict__ Wb,   // [Wq | Wk | Wv] bf16
    bf16_t* __restrict__ Qws, bf16_t* __restrict__ Kws, bf16_t* __restrict__ Vt)
{
    const int lane = threadIdx.x & 31;
    const int wid  = threadIdx.x >> 5;
    const int m0   = (blockIdx.x * 4 + wid) * 16;   // tile base row
    const int half = lane >> 4;                     // lane half (0/1)
    const int ln   = lane & 15;

    const bf16_t* Wqb = Wb;
    const bf16_t* Wkb = Wb + WSZ;
    const bf16_t* Wvb = Wb + 2 * WSZ;

    v8f accq[4] = {}, acck[4] = {}, accv[4] = {};

    const int arow = m0 + ln;                       // A-matrix: row M = lane%16
    for (int kk = 0; kk < DD / 32; ++kk) {
        // A fragment (16x32 bf16): elems 0..15 -> K=half*16+e
        const float* xp = X + (size_t)arow * DD + kk * 32 + half * 16;
        v16bf a;
        #pragma unroll
        for (int e = 0; e < 8; ++e) a[e]     = (bf16_t)xp[e];
        #pragma unroll
        for (int e = 0; e < 8; ++e) a[8 + e] = (bf16_t)xp[8 + e];

        #pragma unroll
        for (int nt = 0; nt < 4; ++nt) {
            // B fragment (32x16): lane = column N = output h, K = half*16 + e.
            // B[k,n] = W[n,k] -> 16 contiguous bf16 = one aligned 32B load.
            const size_t wo = (size_t)(nt * 16 + ln) * DD + kk * 32 + half * 16;
            v16bf bq = *(const v16bf*)(Wqb + wo);
            v16bf bk = *(const v16bf*)(Wkb + wo);
            v16bf bv = *(const v16bf*)(Wvb + wo);
            accq[nt] = __builtin_amdgcn_wmma_f32_16x16x32_bf16(false, a, false, bq, (short)0, accq[nt], false, false);
            acck[nt] = __builtin_amdgcn_wmma_f32_16x16x32_bf16(false, a, false, bk, (short)0, acck[nt], false, false);
            accv[nt] = __builtin_amdgcn_wmma_f32_16x16x32_bf16(false, a, false, bv, (short)0, accv[nt], false, false);
        }
    }

    const float qscale = 0.03125f;  // 1/sqrt(1024)
    #pragma unroll
    for (int nt = 0; nt < 4; ++nt) {
        #pragma unroll
        for (int r = 0; r < 8; ++r) {
            // C layout: VGPR r -> row half*8+r, lane%16 -> column
            const int row = m0 + half * 8 + r;          // global row in [0,16384)
            const int h   = nt * 16 + ln;
            const size_t idx = (size_t)row * HS + h;
            Qws[idx] = (bf16_t)(accq[nt][r] * qscale);
            Kws[idx] = (bf16_t)(acck[nt][r]);
            // V transposed per batch: Vt[b][h][s]
            const int b    = row >> 11;                 // row / 2048
            const int srow = row & (SEQ - 1);
            Vt[(size_t)b * HS * SEQ + (size_t)h * SEQ + srow] = (bf16_t)(accv[nt][r]);
        }
    }
}

// ---------------------------------------------------------------------------
// TDM: DMA a 2-D bf16 tile into LDS.  Descriptor packing per CDNA5 ISA
// §8.3/8.4 (groups 2/3 zero).  6-arg builtin (clang-23 / therock-10.0).
//   tile_d0 x tile_d1 tile from a tensor with row length tensor_d0,
//   row count tensor_d1, row stride stride0 (all in elements).
// ---------------------------------------------------------------------------
__device__ __forceinline__ void tdm_load_2d(unsigned lds_addr, const void* gptr,
                                            unsigned tile_d0, unsigned tile_d1,
                                            unsigned tensor_d0, unsigned tensor_d1,
                                            unsigned stride0)
{
#if defined(__HIP_DEVICE_COMPILE__)
    const unsigned long long ga = (unsigned long long)(uintptr_t)gptr;
    u32x4 g0;
    g0[0] = 1u;                                        // count=1, user descriptor
    g0[1] = lds_addr;                                  // lds_addr [63:32]
    g0[2] = (unsigned)(ga & 0xFFFFFFFFu);              // global_addr [95:64]
    g0[3] = (unsigned)((ga >> 32) & 0x1FFFFFFu)        // global_addr [120:96]
          | (2u << 30);                                // type=2 ("image")
    i32x8 g1;
    g1[0] = (int)(1u << 16);                           // data_size=1 -> 2 bytes
    g1[1] = (int)((tensor_d0 & 0xFFFFu) << 16);        // tensor_dim0 lo16 (63:48)
    g1[2] = (int)(((tensor_d0 >> 16) & 0xFFFFu)        // tensor_dim0 hi16 (79:64)
                | ((tensor_d1 & 0xFFFFu) << 16));      // tensor_dim1 lo16 (95:80)
    g1[3] = (int)(((tensor_d1 >> 16) & 0xFFFFu)        // tensor_dim1 hi16 (111:96)
                | ((tile_d0 & 0xFFFFu) << 16));        // tile_dim0 (127:112)
    g1[4] = (int)(tile_d1 & 0xFFFFu);                  // tile_dim1 (143:128)
    g1[5] = (int)stride0;                              // tensor_dim0_stride lo32
    g1[6] = 0;                                         // stride hi16 + dim1_stride
    g1[7] = 0;
    i32x4 z4 = {};
    i32x8 z8 = {};
    __builtin_amdgcn_tensor_load_to_lds(g0, g1, z4, z4, z8, 0);
#else
    (void)lds_addr; (void)gptr; (void)tile_d0; (void)tile_d1;
    (void)tensor_d0; (void)tensor_d1; (void)stride0;
#endif
}

__device__ __forceinline__ void wait_tensorcnt_le2() {
#if defined(__HIP_DEVICE_COMPILE__)
    __builtin_amdgcn_s_wait_tensorcnt(2);
#endif
}
__device__ __forceinline__ void wait_tensorcnt_0() {
#if defined(__HIP_DEVICE_COMPILE__)
    __builtin_amdgcn_s_wait_tensorcnt(0);
#endif
}
__device__ __forceinline__ void wait_dscnt_0() {
#if defined(__HIP_DEVICE_COMPILE__)
    asm volatile("s_wait_dscnt 0" ::: "memory");
#endif
}

// ---------------------------------------------------------------------------
// Kernel 2: flash attention.  256 threads = 8 waves; each wave owns a 16-row
// query tile (block covers 128 q rows of one batch).  K (row-major) and V
// (pre-transposed) 64x64 bf16 tiles stream through double-buffered LDS via
// the Tensor Data Mover; all LDS fragment reads are contiguous 32B vectors.
// ---------------------------------------------------------------------------
__global__ __launch_bounds__(256) void attn(
    const bf16_t* __restrict__ Qws,
    const bf16_t* __restrict__ Kws,
    const bf16_t* __restrict__ Vt,
    float* __restrict__ Out)
{
    __shared__ __align__(128) bf16_t sK[2][64 * 64];   // [key][h]
    __shared__ __align__(128) bf16_t sV[2][64 * 64];   // [h][key]  (transposed)
    __shared__ __align__(128) bf16_t sP[8][16 * 64];

    const int tid  = threadIdx.x;
    const int lane = tid & 31;
    const int wid  = tid >> 5;
    const int half = lane >> 4;
    const int ln   = lane & 15;

    const int b   = blockIdx.x >> 4;          // batch
    const int qb  = blockIdx.x & 15;          // 128-row query block
    const int q0  = qb * 128;
    const int qr0 = q0 + wid * 16;            // this wave's query rows
    const int nkt = qb * 2 + 2;               // key tiles needed (causal)
    const size_t base  = (size_t)b * SEQ * HS;
    const size_t vbase = (size_t)b * HS * SEQ;

    // Q fragments (A-layout), loaded once from global.
    v16bf aq[2];
    {
        const bf16_t* qp = Qws + base + (size_t)(qr0 + ln) * HS;
        #pragma unroll
        for (int kc = 0; kc < 2; ++kc) {
            const bf16_t* p0 = qp + kc * 32 + half * 16;
            #pragma unroll
            for (int e = 0; e < 8; ++e) aq[kc][e]     = p0[e];
            #pragma unroll
            for (int e = 0; e < 8; ++e) aq[kc][8 + e] = p0[8 + e];
        }
    }

    v8f o[4] = {};
    float m[8], l[8];
    #pragma unroll
    for (int r = 0; r < 8; ++r) { m[r] = -1e30f; l[r] = 0.f; }

    // Prologue: stage tile 0 into buffer 0 (one TDM issue per block: wave 0).
    if (wid == 0) {
        tdm_load_2d((unsigned)(uintptr_t)&sK[0][0], Kws + base, 64, 64, HS, SEQ, HS);
        tdm_load_2d((unsigned)(uintptr_t)&sV[0][0], Vt + vbase, 64, 64, SEQ, HS, SEQ);
    }

    for (int kt = 0; kt < nkt; ++kt) {
        const int buf = kt & 1;
        if (wid == 0) {
            if (kt + 1 < nkt) {           // prefetch next tile, then wait current
                const int nb2 = (kt + 1) & 1;
                tdm_load_2d((unsigned)(uintptr_t)&sK[nb2][0],
                            Kws + base + (size_t)(kt + 1) * 64 * HS, 64, 64, HS, SEQ, HS);
                tdm_load_2d((unsigned)(uintptr_t)&sV[nb2][0],
                            Vt + vbase + (size_t)(kt + 1) * 64, 64, 64, SEQ, HS, SEQ);
                wait_tensorcnt_le2();     // first 2 (current tile) complete
            } else {
                wait_tensorcnt_0();
            }
        }
        __syncthreads();

        const int kb = kt * 64;
        if (kb <= qr0 + 15) {             // causal: tile intersects this wave
            // ---- S = Q @ K^T (scale already folded into Q) ----
            v8f c[4] = {};
            #pragma unroll
            for (int nt = 0; nt < 4; ++nt) {
                #pragma unroll
                for (int kc = 0; kc < 2; ++kc) {
                    const bf16_t* kp = &sK[buf][(nt * 16 + ln) * HS + kc * 32 + half * 16];
                    v16bf bk = *(const v16bf*)kp;
                    c[nt] = __builtin_amdgcn_wmma_f32_16x16x32_bf16(false, aq[kc], false, bk, (short)0, c[nt], false, false);
                }
            }
            // ---- causal mask ----
            #pragma unroll
            for (int nt = 0; nt < 4; ++nt) {
                const int key = kb + nt * 16 + ln;
                #pragma unroll
                for (int r = 0; r < 8; ++r)
                    if (key > qr0 + half * 8 + r) c[nt][r] = -1e30f;
            }
            // ---- online softmax (row lives in 16 lanes of this half-wave) ----
            #pragma unroll
            for (int r = 0; r < 8; ++r) {
                float mx = fmaxf(fmaxf(c[0][r], c[1][r]), fmaxf(c[2][r], c[3][r]));
                mx = fmaxf(mx, __shfl_xor(mx, 1));
                mx = fmaxf(mx, __shfl_xor(mx, 2));
                mx = fmaxf(mx, __shfl_xor(mx, 4));
                mx = fmaxf(mx, __shfl_xor(mx, 8));
                const float mnew  = fmaxf(m[r], mx);
                const float alpha = __expf(m[r] - mnew);
                float rs = 0.f;
                #pragma unroll
                for (int nt = 0; nt < 4; ++nt) {
                    const float p = __expf(c[nt][r] - mnew);
                    c[nt][r] = p;
                    rs += p;
                }
                rs += __shfl_xor(rs, 1);
                rs += __shfl_xor(rs, 2);
                rs += __shfl_xor(rs, 4);
                rs += __shfl_xor(rs, 8);
                l[r] = l[r] * alpha + rs;
                m[r] = mnew;
                #pragma unroll
                for (int ht = 0; ht < 4; ++ht) o[ht][r] *= alpha;
            }
            // ---- P: C-layout -> A-layout via per-wave LDS scratch ----
            #pragma unroll
            for (int nt = 0; nt < 4; ++nt)
                #pragma unroll
                for (int r = 0; r < 8; ++r)
                    sP[wid][(half * 8 + r) * 64 + nt * 16 + ln] = (bf16_t)c[nt][r];
            wait_dscnt_0();               // same-wave DS is in-order; belt & braces
            v16bf ap[2];
            #pragma unroll
            for (int kc = 0; kc < 2; ++kc) {
                const bf16_t* pp = &sP[wid][ln * 64 + kc * 32 + half * 16];
                #pragma unroll
                for (int e = 0; e < 8; ++e) ap[kc][e]     = pp[e];
                #pragma unroll
                for (int e = 0; e < 8; ++e) ap[kc][8 + e] = pp[8 + e];
            }
            // ---- O += P @ V  (V tile transposed: contiguous 32B B-fragments) ----
            #pragma unroll
            for (int ht = 0; ht < 4; ++ht) {
                #pragma unroll
                for (int kc = 0; kc < 2; ++kc) {
                    const bf16_t* vp = &sV[buf][(ht * 16 + ln) * 64 + kc * 32 + half * 16];
                    v16bf bv = *(const v16bf*)vp;
                    o[ht] = __builtin_amdgcn_wmma_f32_16x16x32_bf16(false, ap[kc], false, bv, (short)0, o[ht], false, false);
                }
            }
        }
        __syncthreads();                  // buffer reuse fence
    }

    // ---- normalize and write [B,S,H] f32 ----
    #pragma unroll
    for (int r = 0; r < 8; ++r) {
        const float inv = 1.0f / l[r];
        const size_t row = base + (size_t)(qr0 + half * 8 + r) * HS;
        #pragma unroll
        for (int ht = 0; ht < 4; ++ht)
            Out[row + ht * 16 + ln] = o[ht][r] * inv;
    }
}

// ---------------------------------------------------------------------------
extern "C" void kernel_launch(void* const* d_in, const int* in_sizes, int n_in,
                              void* d_out, int out_size, void* d_ws, size_t ws_size,
                              hipStream_t stream)
{
    const float* X  = (const float*)d_in[0];
    const float* Wk = (const float*)d_in[1];
    const float* Wq = (const float*)d_in[2];
    const float* Wv = (const float*)d_in[3];

    bf16_t* Qws = (bf16_t*)d_ws;                       // 16384*64 bf16 = 2 MB
    bf16_t* Kws = Qws + (size_t)BS * HS;               // 2 MB
    bf16_t* Vt  = Kws + (size_t)BS * HS;               // 2 MB (transposed per batch)
    bf16_t* Wb  = Vt  + (size_t)BS * HS;               // 3*128 KB bf16 weights

    wcvt<<<dim3(WSZ / 256), dim3(256), 0, stream>>>(Wk, Wq, Wv, Wb);
    qkv_proj<<<dim3(BS / 64), dim3(128), 0, stream>>>(X, Wb, Qws, Kws, Vt);
    attn<<<dim3(NB * (SEQ / 128)), dim3(256), 0, stream>>>(Qws, Kws, Vt, (float*)d_out);
}
